// Attention_30494267801907
// MI455X (gfx1250) — matrix operand order repaired and checked by reading, not compiled, not run on hardware
//
#include <hip/hip_runtime.h>
#include <hip/hip_bf16.h>

#define N_CTX 4096
#define NHEADS 16
#define DHEAD 64
#define DIM 1024

typedef _Float16 h2  __attribute__((ext_vector_type(2)));
typedef _Float16 h8  __attribute__((ext_vector_type(8)));
typedef _Float16 h16 __attribute__((ext_vector_type(16)));
typedef float    f8  __attribute__((ext_vector_type(8)));

// ---------------- WMMA fragment helpers (rows/cols per ISA 7.12.2; k order contiguous in A and B) ----------------

// A 16x32 f16: lane m(<16)=row m, elems 0..15 = K k0..k0+15 contiguous
//              lane m+16: K k0+16..k0+31. Two 16B chunks per lane.
static __device__ __forceinline__ h16 load_a_frag(const _Float16* rowp, int k0, int hi) {
  const _Float16* p = rowp + k0 + hi * 16;
  h8 c0 = *(const h8*)(p);
  h8 c1 = *(const h8*)(p + 8);
  return __builtin_shufflevector(c0, c1, 0,1,2,3,4,5,6,7,8,9,10,11,12,13,14,15);
}

// B 32x16 f16: lane n(<16)=col n, elems = K k0..k0+15 contiguous; lane n+16: k0+16..k0+31.
// One 32B contiguous chunk per lane.
static __device__ __forceinline__ h16 load_b_frag(const _Float16* base, int ld, int col0,
                                                  int k0, int lane) {
  const _Float16* p = base + (size_t)(col0 + (lane & 15)) * ld + k0 + ((lane >> 4) << 4);
  return *(const h16*)(p);
}

static __device__ __forceinline__ f8 wmma_f16(h16 a, h16 b, f8 c) {
  return __builtin_amdgcn_wmma_f32_16x16x32_f16(false, a, false, b, (short)0, c, false, false);
}

// ---------------- Stage 1: RoPE + L2 norm + qk_scale, f32 -> f16 head-major ----------------
// One wave per (n, h). Lane i owns the rope pair (2i, 2i+1). Output [h][n][64] f16.
__global__ void __launch_bounds__(128) prep_qk_kernel(const float* __restrict__ x,
                                                      const float* __restrict__ qk_scale,
                                                      _Float16* __restrict__ out16) {
  const int lane = threadIdx.x & 31;
  const int wv   = threadIdx.x >> 5;
  const int idx  = blockIdx.x * 4 + wv;       // n * NHEADS + h
  const int n = idx / NHEADS;
  const int h = idx % NHEADS;
  const int d0 = lane * 2;

  const float xr = x[(size_t)n * DIM + h * DHEAD + d0];
  const float xi = x[(size_t)n * DIM + h * DHEAD + d0 + 1];

  // inv_freq = 10000^(-2i/64) = 2^(-log2(10000) * i/32)
  const float inv_f = exp2f(-13.287712379549449f * ((float)lane * (1.0f / 32.0f)));
  const float ang = (float)n * inv_f;
  float s, c;
  __sincosf(ang, &s, &c);
  const float orr = xr * c - xi * s;
  const float oi  = xr * s + xi * c;

  float ss = orr * orr + oi * oi;
  #pragma unroll
  for (int m = 16; m >= 1; m >>= 1) ss += __shfl_xor(ss, m, 32);
  const float inv = 1.0f / fmaxf(sqrtf(ss), 1e-12f);

  h2 o;
  o.x = (_Float16)(orr * inv * qk_scale[d0]);
  o.y = (_Float16)(oi  * inv * qk_scale[d0 + 1]);
  *(h2*)&out16[((size_t)h * N_CTX + n) * DHEAD + d0] = o;
}

// ---------------- Stage 1b: V transpose [n][h*64+d] f32 -> [h*64+d][n] f16 ----------------
__global__ void prep_vt_kernel(const float* __restrict__ v, _Float16* __restrict__ vt) {
  __shared__ float tile[32][33];
  const int cb = blockIdx.x * 32;  // column index c = h*64 + d
  const int nb = blockIdx.y * 32;  // sequence index n
  const int tx = threadIdx.x, ty = threadIdx.y;  // 32 x 8
  #pragma unroll
  for (int j = 0; j < 32; j += 8)
    tile[ty + j][tx] = v[(size_t)(nb + ty + j) * DIM + cb + tx];
  __syncthreads();
  #pragma unroll
  for (int j = 0; j < 32; j += 8)
    vt[(size_t)(cb + ty + j) * N_CTX + nb + tx] = (_Float16)tile[tx][ty + j];
}

// ---------------- Stage 1c: w_out f32 -> f16 ----------------
__global__ void prep_w_kernel(const float* __restrict__ w, _Float16* __restrict__ w16, int n) {
  int i = blockIdx.x * blockDim.x + threadIdx.x;
  if (i < n) w16[i] = (_Float16)w[i];
}

// ---------------- Stage 2: flash attention ----------------
// Block = 4 waves; wave w owns 32 query rows (two 16-row A blocks) of head h.
// Per 64-wide K/V tile: 16 QK^T WMMAs + 16 PV WMMAs + 4 row-sum WMMAs (l = P @ ones),
// sharing one set of 8 K-fragments and 8 V-fragments across both row blocks.
__global__ void __launch_bounds__(128) flash_attn_kernel(const _Float16* __restrict__ q16,
                                                         const _Float16* __restrict__ k16,
                                                         const _Float16* __restrict__ vt16,
                                                         _Float16* __restrict__ attn16) {
  __shared__ _Float16 pbuf[4][32][72];  // per-wave P transpose staging (padded)
  const int lane = threadIdx.x & 31;
  const int w    = threadIdx.x >> 5;
  const int hi   = lane >> 4;
  const int ln   = lane & 15;
  const int h    = blockIdx.y;
  const int rbase = blockIdx.x * 128 + w * 32;

  const _Float16* qh = q16  + (size_t)h * N_CTX * DHEAD;
  const _Float16* kh = k16  + (size_t)h * N_CTX * DHEAD;
  const _Float16* vh = vt16 + (size_t)h * DHEAD * N_CTX;

  // Q A-fragments stay resident in registers for the whole kernel
  h16 qa[2][2];
  #pragma unroll
  for (int rb = 0; rb < 2; ++rb) {
    const _Float16* qrow = qh + (size_t)(rbase + rb * 16 + ln) * DHEAD;
    qa[rb][0] = load_a_frag(qrow, 0, hi);
    qa[rb][1] = load_a_frag(qrow, 32, hi);
  }

  float m_i[2][8];
  f8 acc[2][4];   // O accumulator fragments
  f8 lfrag[2];    // running softmax denominator as a C/D fragment (all cols = rowsum)
  #pragma unroll
  for (int rb = 0; rb < 2; ++rb) {
    lfrag[rb] = (f8)(0.0f);
    #pragma unroll
    for (int r = 0; r < 8; ++r) m_i[rb][r] = -1e30f;
    #pragma unroll
    for (int f = 0; f < 4; ++f) acc[rb][f] = (f8)(0.0f);
  }

  const h16 ones = (h16)((_Float16)1.0f);
  const float cl = 0.125f * 1.4426950408889634f;  // D^-0.5 * log2(e)

  for (int jb = 0; jb < N_CTX; jb += 64) {
    if (jb + 64 < N_CTX) {  // global_prefetch of next K/V tiles
      __builtin_prefetch(kh + (size_t)(jb + 64 + lane) * DHEAD, 0, 3);
      __builtin_prefetch(vh + (size_t)lane * N_CTX + jb + 64, 0, 3);
    }

    // ---- K tile B-fragments, shared by both row blocks ----
    h16 kb[4][2];
    #pragma unroll
    for (int nb = 0; nb < 4; ++nb) {
      kb[nb][0] = load_b_frag(kh, DHEAD, jb + nb * 16, 0, lane);
      kb[nb][1] = load_b_frag(kh, DHEAD, jb + nb * 16, 32, lane);
    }

    // ---- scores + online max + exp -> LDS (per row block) ----
    #pragma unroll
    for (int rb = 0; rb < 2; ++rb) {
      f8 s[4];
      #pragma unroll
      for (int nb = 0; nb < 4; ++nb) {
        f8 c = (f8)(0.0f);
        c = wmma_f16(qa[rb][0], kb[nb][0], c);
        c = wmma_f16(qa[rb][1], kb[nb][1], c);
        s[nb] = c;
      }
      #pragma unroll
      for (int r = 0; r < 8; ++r) {
        float mr = fmaxf(fmaxf(s[0][r], s[1][r]), fmaxf(s[2][r], s[3][r]));
        #pragma unroll
        for (int msk = 8; msk >= 1; msk >>= 1) mr = fmaxf(mr, __shfl_xor(mr, msk, 32));
        const float mn = fmaxf(m_i[rb][r], mr);
        const float alpha = exp2f((m_i[rb][r] - mn) * cl);
        m_i[rb][r] = mn;
        lfrag[rb][r] *= alpha;
        #pragma unroll
        for (int f = 0; f < 4; ++f) acc[rb][f][r] *= alpha;
        #pragma unroll
        for (int nb = 0; nb < 4; ++nb)
          pbuf[w][rb * 16 + r + hi * 8][nb * 16 + ln] =
              (_Float16)exp2f((s[nb][r] - mn) * cl);
      }
    }

#if __has_builtin(__builtin_amdgcn_s_wait_dscnt)
    __builtin_amdgcn_s_wait_dscnt(0);
#endif
    // ---- reload P in A layout ----
    h16 pa[2][2];
    #pragma unroll
    for (int rb = 0; rb < 2; ++rb) {
      pa[rb][0] = load_a_frag(&pbuf[w][rb * 16 + ln][0], 0, hi);
      pa[rb][1] = load_a_frag(&pbuf[w][rb * 16 + ln][0], 32, hi);
    }

    // ---- row sums on the matrix pipe: l += P @ ones ----
    #pragma unroll
    for (int rb = 0; rb < 2; ++rb) {
      lfrag[rb] = wmma_f16(pa[rb][0], ones, lfrag[rb]);
      lfrag[rb] = wmma_f16(pa[rb][1], ones, lfrag[rb]);
    }

    // ---- O += P V ; V fragments shared by both row blocks ----
    #pragma unroll
    for (int f = 0; f < 4; ++f) {
      h16 vb0 = load_b_frag(vh, N_CTX, f * 16, jb, lane);
      h16 vb1 = load_b_frag(vh, N_CTX, f * 16, jb + 32, lane);
      #pragma unroll
      for (int rb = 0; rb < 2; ++rb) {
        f8 o = acc[rb][f];
        o = wmma_f16(pa[rb][0], vb0, o);
        o = wmma_f16(pa[rb][1], vb1, o);
        acc[rb][f] = o;
      }
    }
  }

  // epilogue: O / l, emit f16 in [n][h*64+d] layout for the projection GEMM
  #pragma unroll
  for (int rb = 0; rb < 2; ++rb)
    #pragma unroll
    for (int r = 0; r < 8; ++r) {
      const float invl = 1.0f / lfrag[rb][r];
      const int row = rbase + rb * 16 + r + hi * 8;
      #pragma unroll
      for (int f = 0; f < 4; ++f)
        attn16[(size_t)row * DIM + h * DHEAD + f * 16 + ln] =
            (_Float16)(acc[rb][f][r] * invl);
    }
}

// ---------------- Stage 3: out = attn @ W^T + b (4096x1024 x 1024x1024) ----------------
// Wave owns 32 rows x 64 cols; both row blocks share each W fragment.
__global__ void __launch_bounds__(128) out_proj_kernel(const _Float16* __restrict__ a16,
                                                       const _Float16* __restrict__ w16,
                                                       const float* __restrict__ bias,
                                                       float* __restrict__ out) {
  const int lane = threadIdx.x & 31;
  const int w    = threadIdx.x >> 5;
  const int hi   = lane >> 4;
  const int ln   = lane & 15;
  const int rbase = blockIdx.x * 128 + w * 32;
  const int cbase = blockIdx.y * 64;

  f8 acc[2][4];
  #pragma unroll
  for (int rb = 0; rb < 2; ++rb)
    #pragma unroll
    for (int f = 0; f < 4; ++f) acc[rb][f] = (f8)(0.0f);

  const _Float16* arow0 = a16 + (size_t)(rbase + ln) * DIM;
  const _Float16* arow1 = a16 + (size_t)(rbase + 16 + ln) * DIM;
  for (int k0 = 0; k0 < DIM; k0 += 32) {
    h16 a0 = load_a_frag(arow0, k0, hi);
    h16 a1 = load_a_frag(arow1, k0, hi);
    #pragma unroll
    for (int f = 0; f < 4; ++f) {
      h16 b = load_b_frag(w16, DIM, cbase + f * 16, k0, lane);
      acc[0][f] = wmma_f16(a0, b, acc[0][f]);
      acc[1][f] = wmma_f16(a1, b, acc[1][f]);
    }
  }
  #pragma unroll
  for (int f = 0; f < 4; ++f) {
    const int col = cbase + f * 16 + ln;
    const float b = bias[col];
    #pragma unroll
    for (int rb = 0; rb < 2; ++rb)
      #pragma unroll
      for (int r = 0; r < 8; ++r)
        out[(size_t)(rbase + rb * 16 + r + hi * 8) * DIM + col] = acc[rb][f][r] + b;
  }
}

// ---------------- host ----------------
extern "C" void kernel_launch(void* const* d_in, const int* in_sizes, int n_in,
                              void* d_out, int out_size, void* d_ws, size_t ws_size,
                              hipStream_t stream) {
  const float* q        = (const float*)d_in[0];
  const float* k        = (const float*)d_in[1];
  const float* v        = (const float*)d_in[2];
  const float* qk_scale = (const float*)d_in[3];
  const float* w_out    = (const float*)d_in[4];
  const float* b_out    = (const float*)d_in[5];
  float* out = (float*)d_out;

  char* ws = (char*)d_ws;
  const size_t sz = (size_t)NHEADS * N_CTX * DHEAD * sizeof(_Float16);  // 8 MB
  _Float16* q16    = (_Float16*)(ws);
  _Float16* k16    = (_Float16*)(ws + sz);
  _Float16* vt16   = (_Float16*)(ws + 2 * sz);
  _Float16* attn16 = (_Float16*)(ws + 3 * sz);
  _Float16* w16    = (_Float16*)(ws + 4 * sz);

  prep_qk_kernel<<<dim3(N_CTX * NHEADS / 4), 128, 0, stream>>>(q, qk_scale, q16);
  prep_qk_kernel<<<dim3(N_CTX * NHEADS / 4), 128, 0, stream>>>(k, qk_scale, k16);
  prep_vt_kernel<<<dim3(DIM / 32, N_CTX / 32), dim3(32, 8), 0, stream>>>(v, vt16);
  prep_w_kernel<<<dim3((DIM * DIM + 255) / 256), 256, 0, stream>>>(w_out, w16, DIM * DIM);
  flash_attn_kernel<<<dim3(N_CTX / 128, NHEADS), 128, 0, stream>>>(q16, k16, vt16, attn16);
  out_proj_kernel<<<dim3(N_CTX / 128, DIM / 64), 128, 0, stream>>>(attn16, w16, b_out, out);
}
